// VQVAE_58248346468915
// MI455X (gfx1250) — matrix operand order repaired and checked by reading, not compiled, not run on hardware
//
#include <hip/hip_runtime.h>
#include <hip/hip_bf16.h>
#include <math.h>

typedef __bf16 bf16;
typedef __attribute__((ext_vector_type(16))) __bf16 v16bf;
typedef __attribute__((ext_vector_type(8)))  __bf16 v8bf;
typedef __attribute__((ext_vector_type(8)))  float  v8f;

__device__ __forceinline__ bf16 cvt_bf16(float x) { return (bf16)x; }
__device__ __forceinline__ bf16 cvt_bf16(bf16 x)  { return x; }

__device__ __forceinline__ v16bf combine16(v8bf lo, v8bf hi) {
  v16bf r;
#pragma unroll
  for (int i = 0; i < 8; ++i) { r[i] = lo[i]; r[i + 8] = hi[i]; }
  return r;
}

// A fragment (16x32 bf16): p -> start of this lane's row's 32-wide K-slice.
// lane<16: K {0..7,16..23}; lane>=16: K {8..15,24..31}  (ISA 7.12.2)
__device__ __forceinline__ v16bf afrag(const bf16* p, int lane) {
  int ko = (lane >> 4) << 3;                     // 0 or 8
  v8bf lo = *(const v8bf*)(p + ko);
  v8bf hi = *(const v8bf*)(p + ko + 16);
  return combine16(lo, hi);
}

// B fragment (32x16 bf16), B staged transposed [N][K]: p -> start of column
// (lane&15)'s 32-wide K-slice.  lane<16: K {0..7,16..23}; lane>=16: K {8..15,24..31}
__device__ __forceinline__ v16bf bfrag(const bf16* p, int lane) {
  int kb = (lane >> 4) << 3;                     // 0 or 8
  v8bf lo = *(const v8bf*)(p + kb);
  v8bf hi = *(const v8bf*)(p + kb + 16);
  return combine16(lo, hi);
}

// ---------------------------------------------------------------------------
// out[M][64] (bf16) = act(A[M][K] @ W[K][64] + bias), optional ReLU.
// Block tile 32(M) x 64(N), K-step 64, register-staged double-buffered LDS.
// 8 waves: (wm:2)x(wn:4), each owns a 16x16 tile, 2 WMMAs per K-step.
// ---------------------------------------------------------------------------
template <typename AT>
__global__ __launch_bounds__(256) void gemm_n64_kernel(
    const AT* __restrict__ A, const float* __restrict__ W,
    const float* __restrict__ bias, bf16* __restrict__ out,
    int M, int K, int relu) {
  __shared__ bf16 As[2][32][64];                  // 2 x 4 KB
  __shared__ bf16 Bt[2][64][64];                  // 2 x 8 KB
  const int t = threadIdx.x, lane = t & 31, wave = t >> 5;
  const int wm = wave >> 2, wn = wave & 3;
  const int m0 = blockIdx.x * 32;
  const int nsteps = K >> 6;

  AT    aReg[8];
  float bReg[16];

  // prologue: stage step 0
#pragma unroll
  for (int i = 0; i < 8; ++i) {
    int idx = t + i * 256;
    aReg[i] = A[(size_t)(m0 + (idx >> 6)) * K + (idx & 63)];
  }
#pragma unroll
  for (int i = 0; i < 16; ++i) {
    int idx = t + i * 256;
    bReg[i] = W[(size_t)(idx >> 6) * 64 + (idx & 63)];
  }
#pragma unroll
  for (int i = 0; i < 8; ++i) {
    int idx = t + i * 256;
    As[0][idx >> 6][idx & 63] = cvt_bf16(aReg[i]);
  }
#pragma unroll
  for (int i = 0; i < 16; ++i) {
    int idx = t + i * 256;
    Bt[0][idx & 63][idx >> 6] = cvt_bf16(bReg[i]);
  }
  __syncthreads();

  v8f acc = {};
  for (int s = 0; s < nsteps; ++s) {
    const int cur = s & 1;
    const bool more = (s + 1) < nsteps;
    if (more) {                                   // prefetch next tile to regs
      const int kk = (s + 1) << 6;
#pragma unroll
      for (int i = 0; i < 8; ++i) {
        int idx = t + i * 256;
        aReg[i] = A[(size_t)(m0 + (idx >> 6)) * K + kk + (idx & 63)];
      }
#pragma unroll
      for (int i = 0; i < 16; ++i) {
        int idx = t + i * 256;
        bReg[i] = W[(size_t)(kk + (idx >> 6)) * 64 + (idx & 63)];
      }
    }
    // compute on current buffer while prefetch loads are in flight
    const bf16* arow = &As[cur][wm * 16 + (lane & 15)][0];
    const bf16* brow = &Bt[cur][wn * 16 + (lane & 15)][0];
#pragma unroll
    for (int h = 0; h < 2; ++h) {
      v16bf a = afrag(arow + h * 32, lane);
      v16bf b = bfrag(brow + h * 32, lane);
      acc = __builtin_amdgcn_wmma_f32_16x16x32_bf16(false, a, false, b,
                                                    (short)0, acc, false, false);
    }
    if (more) {                                   // commit prefetch to LDS
      const int nxt = cur ^ 1;
#pragma unroll
      for (int i = 0; i < 8; ++i) {
        int idx = t + i * 256;
        As[nxt][idx >> 6][idx & 63] = cvt_bf16(aReg[i]);
      }
#pragma unroll
      for (int i = 0; i < 16; ++i) {
        int idx = t + i * 256;
        Bt[nxt][idx & 63][idx >> 6] = cvt_bf16(bReg[i]);
      }
    }
    __syncthreads();
  }

  const int col = wn * 16 + (lane & 15);
  const int rbase = m0 + wm * 16 + ((lane >= 16) ? 8 : 0);
  const float bv = bias[col];
#pragma unroll
  for (int v = 0; v < 8; ++v) {
    float x = acc[v] + bv;
    if (relu) x = fmaxf(x, 0.0f);
    out[(size_t)(rbase + v) * 64 + col] = (bf16)x;
  }
}

// ---------------------------------------------------------------------------
// out[M][N] (f32) = A[M][64](bf16) @ W[64][N] + bias.  Block tile 64x64.
// ---------------------------------------------------------------------------
__global__ __launch_bounds__(256) void gemm_k64_kernel(
    const bf16* __restrict__ A, const float* __restrict__ W,
    const float* __restrict__ bias, float* __restrict__ out, int M, int N) {
  __shared__ bf16 As[64][64];
  __shared__ bf16 Bt[64][64];
  const int t = threadIdx.x, lane = t & 31, wave = t >> 5;
  const int wm = wave >> 1, wn = wave & 1;
  const int m0 = blockIdx.y * 64, n0 = blockIdx.x * 64;

#pragma unroll
  for (int i = 0; i < 16; ++i) {                  // stage A 64x64
    int idx = t + i * 256, r = idx >> 6, c = idx & 63;
    As[r][c] = A[(size_t)(m0 + r) * 64 + c];
  }
#pragma unroll
  for (int i = 0; i < 16; ++i) {                  // stage B^T 64x64
    int idx = t + i * 256, n = idx & 63, k = idx >> 6;
    Bt[n][k] = cvt_bf16(W[(size_t)k * N + n0 + n]);
  }
  __syncthreads();

  v8f acc0 = {}, acc1 = {};
  const bf16* arow  = &As[wm * 16 + (lane & 15)][0];
  const bf16* b0row = &Bt[wn * 32 + (lane & 15)][0];
  const bf16* b1row = &Bt[wn * 32 + 16 + (lane & 15)][0];
#pragma unroll
  for (int h = 0; h < 2; ++h) {
    v16bf a  = afrag(arow + h * 32, lane);
    v16bf b0 = bfrag(b0row + h * 32, lane);
    v16bf b1 = bfrag(b1row + h * 32, lane);
    acc0 = __builtin_amdgcn_wmma_f32_16x16x32_bf16(false, a, false, b0,
                                                   (short)0, acc0, false, false);
    acc1 = __builtin_amdgcn_wmma_f32_16x16x32_bf16(false, a, false, b1,
                                                   (short)0, acc1, false, false);
  }
  const int rbase = m0 + wm * 16 + ((lane >= 16) ? 8 : 0);
  const int c0 = n0 + wn * 32 + (lane & 15), c1 = c0 + 16;
  const float bv0 = bias[c0], bv1 = bias[c1];
#pragma unroll
  for (int v = 0; v < 8; ++v) {
    out[(size_t)(rbase + v) * N + c0] = acc0[v] + bv0;
    out[(size_t)(rbase + v) * N + c1] = acc1[v] + bv1;
  }
}

// ---------------------------------------------------------------------------
// VQ: argmin_k ||z - emb_k||^2 via WMMA scores; gather q (bf16), idx (float),
// accumulate sum((q-z)^2).  256 rows/block (16 waves), emb bf16 in LDS.
// ---------------------------------------------------------------------------
__global__ __launch_bounds__(512) void vq_kernel(
    const float* __restrict__ z, const float* __restrict__ emb,
    bf16* __restrict__ q_out, float* __restrict__ idx_out,
    float* __restrict__ loss_acc) {
  extern __shared__ char smem[];
  bf16*  embL = (bf16*)smem;                      // 1024*64 bf16 = 128 KB
  float* e2   = (float*)(smem + 131072);          // 1024 f32   -> 135168
  bf16*  zt   = (bf16*)(smem + 135168);           // 256*64 bf16 -> 167936
  float* z2s  = (float*)(smem + 167936);          // 256 f32    -> 168960
  int*   kid  = (int*)(smem + 168960);            // 256 i32    -> 169984

  const int t = threadIdx.x, lane = t & 31, wave = t >> 5;   // wave 0..15

  for (int i = t; i < 1024 * 64; i += 512) embL[i] = (bf16)emb[i];
  for (int k = t; k < 1024; k += 512) {
    float s = 0.f;
#pragma unroll 8
    for (int e = 0; e < 64; ++e) { float v = emb[k * 64 + e]; s += v * v; }
    e2[k] = s;
  }
  const int rowbase = blockIdx.x * 256 + wave * 16;
  {
    const int r = lane >> 1, half = lane & 1;     // 2 lanes per row
    const float* zr = z + (size_t)(rowbase + r) * 64 + half * 32;
    bf16* zd = zt + (wave * 16 + r) * 64 + half * 32;
    float s = 0.f;
#pragma unroll
    for (int j = 0; j < 32; ++j) { float v = zr[j]; s += v * v; zd[j] = (bf16)v; }
    s += __shfl_xor(s, 1, 32);
    if (half == 0) z2s[wave * 16 + r] = s;
  }
  __syncthreads();

  const bf16* zrow = zt + (wave * 16 + (lane & 15)) * 64;
  v16bf a0 = afrag(zrow, lane);
  v16bf a1 = afrag(zrow + 32, lane);

  float z2v[8];
  {
    int rb = wave * 16 + ((lane >= 16) ? 8 : 0);
#pragma unroll
    for (int v = 0; v < 8; ++v) z2v[v] = z2s[rb + v];
  }
  float bestd[8]; int bestk[8];
#pragma unroll
  for (int v = 0; v < 8; ++v) { bestd[v] = 3.0e38f; bestk[v] = 0; }

  for (int nt = 0; nt < 64; ++nt) {
    const int n = nt * 16 + (lane & 15);
    const bf16* brow = embL + n * 64;
    v16bf b0 = bfrag(brow, lane);
    v16bf b1 = bfrag(brow + 32, lane);
    v8f c = {};
    c = __builtin_amdgcn_wmma_f32_16x16x32_bf16(false, a0, false, b0,
                                                (short)0, c, false, false);
    c = __builtin_amdgcn_wmma_f32_16x16x32_bf16(false, a1, false, b1,
                                                (short)0, c, false, false);
    const float e2n = e2[n];
#pragma unroll
    for (int v = 0; v < 8; ++v) {
      float d = z2v[v] + e2n - 2.0f * c[v];
      if (d < bestd[v]) { bestd[v] = d; bestk[v] = n; }
    }
  }
  // min-reduce over the 16 lanes holding each row; prefer smallest index on tie
#pragma unroll
  for (int v = 0; v < 8; ++v) {
    float d = bestd[v]; int k = bestk[v];
#pragma unroll
    for (int s = 8; s > 0; s >>= 1) {
      float od = __shfl_xor(d, s, 32);
      int   ok = __shfl_xor(k, s, 32);
      if (od < d || (od == d && ok < k)) { d = od; k = ok; }
    }
    if ((lane & 15) == 0) kid[wave * 16 + ((lane >= 16) ? 8 : 0) + v] = k;
  }
  __syncthreads();
  {
    const int r = lane >> 1, half = lane & 1;
    const int k = kid[wave * 16 + r];
    const size_t grow = (size_t)rowbase + r;
    if (half == 0) idx_out[grow] = (float)k;
    const float* er = emb + (size_t)k * 64 + half * 32;
    const float* zr = z + grow * 64 + half * 32;
    bf16* qr = q_out + grow * 64 + half * 32;
    float ls = 0.f;
#pragma unroll
    for (int j = 0; j < 32; ++j) {
      float qv = er[j];
      float df = qv - zr[j];
      ls += df * df;
      qr[j] = (bf16)qv;
    }
#pragma unroll
    for (int s = 16; s > 0; s >>= 1) ls += __shfl_xor(ls, s, 32);
    if (lane == 0) atomicAdd(loss_acc, ls);
  }
}

__global__ void zero_loss_kernel(float* loss_acc) { *loss_acc = 0.f; }

__global__ void finalize_kernel(const float* loss_acc, float* out_scalars) {
  float mse = *loss_acc / 16777216.0f;            // B*L*E
  out_scalars[0] = 1.25f * mse;                   // q_loss + 0.25*e_loss
  float avg = 1.0f / 1024.0f;                     // avg_probs is exactly 1/K
  out_scalars[1] = expf(-(avg * logf(avg + 1e-10f)));
}

extern "C" void kernel_launch(void* const* d_in, const int* in_sizes, int n_in,
                              void* d_out, int out_size, void* d_ws, size_t ws_size,
                              hipStream_t stream) {
  const float* x   = (const float*)d_in[0];
  const float* W1  = (const float*)d_in[1];
  const float* b1  = (const float*)d_in[2];
  const float* W2  = (const float*)d_in[3];
  const float* b2  = (const float*)d_in[4];
  const float* emb = (const float*)d_in[5];
  const float* W3  = (const float*)d_in[6];
  const float* b3  = (const float*)d_in[7];
  const float* W4  = (const float*)d_in[8];
  const float* b4  = (const float*)d_in[9];
  float* out = (float*)d_out;

  char* ws = (char*)d_ws;
  float* z_e  = (float*)(ws);                     // 4096*4096 f32 = 64 MB
  bf16*  h1   = (bf16*)(ws + 67108864);           // 4096*64 bf16
  bf16*  qbf  = (bf16*)(ws + 67633152);           // 4096*4096 bf16 = 32 MB
  bf16*  h2   = (bf16*)(ws + 101187584);          // 4096*64 bf16
  float* lacc = (float*)(ws + 101711872);

  float* x_recon = out;                           // [0, 16777216)
  float* scalars = out + 16777216;                // vq_loss, perplexity
  float* idx_out = out + 16777218;                // 262144 floats

  zero_loss_kernel<<<1, 1, 0, stream>>>(lacc);
  // encoder
  gemm_n64_kernel<float><<<dim3(4096 / 32), 256, 0, stream>>>(
      x, W1, b1, h1, 4096, 4096, 1);
  gemm_k64_kernel<<<dim3(64, 64), 256, 0, stream>>>(h1, W2, b2, z_e, 4096, 4096);
  // vector quantization (argmin + gather + loss)
  vq_kernel<<<dim3(1024), 512, 169984, stream>>>(z_e, emb, qbf, idx_out, lacc);
  // decoder
  gemm_n64_kernel<bf16><<<dim3(4096 / 32), 256, 0, stream>>>(
      qbf, W3, b3, h2, 4096, 4096, 1);
  gemm_k64_kernel<<<dim3(64, 64), 256, 0, stream>>>(h2, W4, b4, x_recon, 4096, 4096);
  finalize_kernel<<<1, 1, 0, stream>>>(lacc, scalars);
}
